// AttentionMAE_7318624272464
// MI455X (gfx1250) — matrix operand order repaired and checked by reading, not compiled, not run on hardware
//
#include <hip/hip_runtime.h>
#include <hip/hip_bf16.h>

typedef __attribute__((ext_vector_type(16))) _Float16 v16h;
typedef __attribute__((ext_vector_type(8)))  _Float16 v8h;
typedef __attribute__((ext_vector_type(8)))  float    v8f;

#define Bc 64
#define Cc 3
#define Hc 224
#define Wc 224
#define PHc 16
#define GHc 14
#define GWc 14
#define NPc 196
#define Ec 512
#define NHc 8
#define HDc 64
#define PPDc 768
#define NMASK 147
#define NUNM 49
#define IMG_ELEMS (Bc*Cc*Hc*Wc)   // 9,633,792

// ---------------------------------------------------------------------------
// WMMA fragment helpers (16x16x32 f16; this code's contiguous k order, same for A and B)
// A (16x32): lane holds row lq, 16 contiguous K starting at kk + (hi?16:0)
// B (32x16): lane holds col lq, 16 contiguous K starting at kk + (hi?16:0)
// ---------------------------------------------------------------------------
__device__ __forceinline__ v16h load_a_frag(const _Float16* row, int kk, bool hi) {
    int ka = kk + (hi ? 16 : 0);
    v8h lo = *(const v8h*)(row + ka);
    v8h hh = *(const v8h*)(row + ka + 8);
    v16h a;
#pragma unroll
    for (int i = 0; i < 8; ++i) { a[i] = lo[i]; a[8 + i] = hh[i]; }
    return a;
}

__device__ __forceinline__ v8f wmma_f16(v16h a, v16h b, v8f c) {
    return __builtin_amdgcn_wmma_f32_16x16x32_f16(false, a, false, b, (short)0, c, false, false);
}

// ---------------------------------------------------------------------------
// Generic WMMA GEMM:  D[M,N] = act[M,K](f16) @ W[N,K](f16)^T + bias
// Each wave: one 32x64 output strip (2 m-tiles x 4 n-tiles). Block = 8 waves.
// M must be a multiple of 32, N a multiple of 64, K a multiple of 32.
// EPI: 0=f16 out, 1=bias+posembed gather -> f16, 2=bias+gelu -> f16,
//      3=bias -> f32, 4=kv split (k f16 row-major, v transposed per head)
// ---------------------------------------------------------------------------
template<int EPI>
__global__ __launch_bounds__(256)
void gemm_wmma(const _Float16* __restrict__ act,
               const _Float16* __restrict__ W,
               const float* __restrict__ bias,
               int M, int N, int K,
               _Float16* __restrict__ out16,
               float* __restrict__ out32,
               const float* __restrict__ pe,
               const int* __restrict__ shuffle,
               int rpb, int idx_off,
               _Float16* __restrict__ k_out,
               _Float16* __restrict__ vT)
{
    const int wave = threadIdx.x >> 5;
    const int lane = threadIdx.x & 31;
    const int ngN  = N >> 6;
    const int group = blockIdx.x * 8 + wave;
    const int mt = group / ngN;
    const int ng = group % ngN;
    if (mt * 32 >= M) return;

    const int  lq = lane & 15;
    const bool hi = lane >= 16;

    const _Float16* arow0 = act + (size_t)(mt * 32 + lq) * K;
    const _Float16* arow1 = arow0 + (size_t)16 * K;
    const _Float16* w0 = W + (size_t)(ng * 64 + 0  + lq) * K;
    const _Float16* w1 = W + (size_t)(ng * 64 + 16 + lq) * K;
    const _Float16* w2 = W + (size_t)(ng * 64 + 32 + lq) * K;
    const _Float16* w3 = W + (size_t)(ng * 64 + 48 + lq) * K;

    v8f c00 = {}, c01 = {}, c02 = {}, c03 = {};
    v8f c10 = {}, c11 = {}, c12 = {}, c13 = {};
    for (int kk = 0; kk < K; kk += 32) {
        v16h a0 = load_a_frag(arow0, kk, hi);
        v16h a1 = load_a_frag(arow1, kk, hi);
        int kb = kk + (hi ? 16 : 0);
        v16h b0 = *(const v16h*)(w0 + kb);
        v16h b1 = *(const v16h*)(w1 + kb);
        v16h b2 = *(const v16h*)(w2 + kb);
        v16h b3 = *(const v16h*)(w3 + kb);
        c00 = wmma_f16(a0, b0, c00);
        c10 = wmma_f16(a1, b0, c10);
        c01 = wmma_f16(a0, b1, c01);
        c11 = wmma_f16(a1, b1, c11);
        c02 = wmma_f16(a0, b2, c02);
        c12 = wmma_f16(a1, b2, c12);
        c03 = wmma_f16(a0, b3, c03);
        c13 = wmma_f16(a1, b3, c13);
    }

#pragma unroll
    for (int mi = 0; mi < 2; ++mi) {
        const int m0 = mt * 32 + mi * 16 + (hi ? 8 : 0);
#pragma unroll
        for (int t = 0; t < 4; ++t) {
            v8f c;
            if (mi == 0) c = (t == 0) ? c00 : (t == 1) ? c01 : (t == 2) ? c02 : c03;
            else         c = (t == 0) ? c10 : (t == 1) ? c11 : (t == 2) ? c12 : c13;
            int n = ng * 64 + t * 16 + lq;
            float bz = bias ? bias[n] : 0.0f;
#pragma unroll
            for (int i = 0; i < 8; ++i) {
                int row = m0 + i;
                float v = c[i] + bz;
                if (EPI == 1) {
                    int bidx = row / rpb, r = row % rpb;
                    int tok = shuffle[bidx * NPc + idx_off + r];
                    v += pe[(size_t)tok * Ec + n];
                    out16[(size_t)row * N + n] = (_Float16)v;
                } else if (EPI == 2) {
                    float u = v + 0.044715f * v * v * v;
                    v = 0.5f * v * (1.0f + tanhf(0.7978845608f * u));
                    out16[(size_t)row * N + n] = (_Float16)v;
                } else if (EPI == 3) {
                    out32[(size_t)row * N + n] = v;
                } else if (EPI == 4) {
                    if (n < Ec) {
                        k_out[(size_t)row * Ec + n] = (_Float16)v;
                    } else {
                        int h = (n - Ec) >> 6, d = (n - Ec) & 63;
                        int bb = row / NUNM, nn = row % NUNM;
                        vT[((((size_t)bb * NHc + h) * HDc) + d) * 64 + nn] = (_Float16)v;
                    }
                } else {
                    out16[(size_t)row * N + n] = (_Float16)v;
                }
            }
        }
    }
}

// ---------------------------------------------------------------------------
// Attention: one workgroup per (b, h). scores = q@k^T/8 -> softmax -> @vT
// Dual accumulators per tile break the WMMA D->C serial chain (no hazard nops).
// ---------------------------------------------------------------------------
__global__ __launch_bounds__(256)
void attention_kernel(const _Float16* __restrict__ q,    // (B*147, 512)
                      const _Float16* __restrict__ kbuf, // (B*49, 512)
                      const _Float16* __restrict__ vT,   // (B,8,64,64) zero-padded n>=49
                      _Float16* __restrict__ attn_out)   // (B*147, 512)
{
    __shared__ float    sc[160 * 64];
    __shared__ _Float16 probs[160 * 64];

    const int bh = blockIdx.x;
    const int b = bh / NHc, h = bh % NHc;
    const int wave = threadIdx.x >> 5;
    const int lane = threadIdx.x & 31;
    const int lq = lane & 15;
    const bool hi = lane >= 16;

    // --- phase 1: scores (10 m-tiles x 4 n-tiles), K = 64 split in 2 chains ---
    for (int tile = wave; tile < 40; tile += 8) {
        int mt = tile >> 2, nt = tile & 3;
        int rowA = mt * 16 + lq; rowA = rowA < NMASK ? rowA : NMASK - 1;
        int nk = nt * 16 + lq;   nk   = nk < NUNM ? nk : NUNM - 1;
        const _Float16* qrow = q    + ((size_t)(b * NMASK + rowA)) * Ec + h * HDc;
        const _Float16* krow = kbuf + ((size_t)(b * NUNM  + nk))   * Ec + h * HDc;
        v8f ca = {}, cb = {};
        {
            v16h a0 = load_a_frag(qrow, 0, hi);
            v16h a1 = load_a_frag(qrow, 32, hi);
            v16h b0 = *(const v16h*)(krow + 0  + (hi ? 16 : 0));
            v16h b1 = *(const v16h*)(krow + 32 + (hi ? 16 : 0));
            ca = wmma_f16(a0, b0, ca);
            cb = wmma_f16(a1, b1, cb);
        }
        int n = nt * 16 + lq;
        int m0 = mt * 16 + (hi ? 8 : 0);
#pragma unroll
        for (int i = 0; i < 8; ++i) sc[(m0 + i) * 64 + n] = (ca[i] + cb[i]) * 0.125f;
    }
    __syncthreads();

    // --- phase 2: row softmax over 49 keys ---
    if (threadIdx.x < 160) {
        int r = threadIdx.x;
        if (r < NMASK) {
            float mx = -1e30f;
            for (int n = 0; n < NUNM; ++n) mx = fmaxf(mx, sc[r * 64 + n]);
            float s = 0.0f;
            for (int n = 0; n < NUNM; ++n) {
                float e = __expf(sc[r * 64 + n] - mx);
                sc[r * 64 + n] = e;
                s += e;
            }
            float inv = 1.0f / s;
            for (int n = 0; n < NUNM; ++n) probs[r * 64 + n] = (_Float16)(sc[r * 64 + n] * inv);
            for (int n = NUNM; n < 64; ++n) probs[r * 64 + n] = (_Float16)0.0f;
        } else {
            for (int n = 0; n < 64; ++n) probs[r * 64 + n] = (_Float16)0.0f;
        }
    }
    __syncthreads();

    // --- phase 3: attn @ v (K = 64 zero-padded keys), dual chains ---
    const _Float16* vbase = vT + ((size_t)(b * NHc + h)) * HDc * 64;
    for (int tile = wave; tile < 40; tile += 8) {
        int mt = tile >> 2, dt = tile & 3;
        const _Float16* prow = &probs[(mt * 16 + lq) * 64];
        const _Float16* vrow = vbase + (size_t)(dt * 16 + lq) * 64;
        v8f ca = {}, cb = {};
        {
            v16h a0 = load_a_frag(prow, 0, hi);
            v16h a1 = load_a_frag(prow, 32, hi);
            v16h b0 = *(const v16h*)(vrow + 0  + (hi ? 16 : 0));
            v16h b1 = *(const v16h*)(vrow + 32 + (hi ? 16 : 0));
            ca = wmma_f16(a0, b0, ca);
            cb = wmma_f16(a1, b1, cb);
        }
        int d = dt * 16 + lq;
        int m0 = mt * 16 + (hi ? 8 : 0);
#pragma unroll
        for (int i = 0; i < 8; ++i) {
            int m = m0 + i;
            if (m < NMASK)
                attn_out[((size_t)(b * NMASK + m)) * Ec + h * HDc + d] = (_Float16)(ca[i] + cb[i]);
        }
    }
}

// ---------------------------------------------------------------------------
// Elementwise kernels
// ---------------------------------------------------------------------------
__global__ void f32_to_f16_kernel(const float* __restrict__ in, _Float16* __restrict__ out, int n) {
    int i = blockIdx.x * blockDim.x + threadIdx.x;
    if (i < n) out[i] = (_Float16)in[i];
}

__global__ void zero_f16_kernel(_Float16* __restrict__ p, int n) {
    int i = blockIdx.x * blockDim.x + threadIdx.x;
    if (i < n) p[i] = (_Float16)0.0f;
}

__global__ void zero_f32_kernel(float* __restrict__ p, int n) {
    int i = blockIdx.x * blockDim.x + threadIdx.x;
    if (i < n) p[i] = 0.0f;
}

__global__ void zero_mask_kernel(unsigned char* __restrict__ m, int n) {
    int i = blockIdx.x * blockDim.x + threadIdx.x;
    if (i < n) m[i] = 0;
}

__global__ void set_mask_kernel(const int* __restrict__ shuffle, unsigned char* __restrict__ m) {
    int i = blockIdx.x * blockDim.x + threadIdx.x;  // over B*147
    if (i < Bc * NMASK) {
        int b = i / NMASK, j = i % NMASK;
        m[b * NPc + shuffle[b * NPc + j]] = 1;
    }
}

__device__ __forceinline__ size_t pix_index(int b, int tok, int d) {
    int gh = tok / GWc, gw = tok % GWc;
    int c = d % Cc, s = d / Cc;
    int pw = s % PHc, ph = s / PHc;
    return (((size_t)(b * Cc + c) * Hc) + gh * PHc + ph) * Wc + gw * PHc + pw;
}

// gather patches for all 196 tokens per image, write f16 to mask/unmask buffers
__global__ void gather_patches_kernel(const float* __restrict__ x,
                                      const int* __restrict__ shuffle,
                                      _Float16* __restrict__ maskp,
                                      _Float16* __restrict__ unmaskp)
{
    size_t i = (size_t)blockIdx.x * blockDim.x + threadIdx.x;
    if (i >= (size_t)Bc * NPc * PPDc) return;
    int d = i % PPDc;
    size_t t = i / PPDc;
    int j = t % NPc;
    int b = t / NPc;
    int tok = shuffle[b * NPc + j];
    float v = x[pix_index(b, tok, d)];
    if (j < NMASK) maskp[((size_t)(b * NMASK + j)) * PPDc + d] = (_Float16)v;
    else           unmaskp[((size_t)(b * NUNM + (j - NMASK))) * PPDc + d] = (_Float16)v;
}

// base images (float4): recons = x, patches_img = x, mask_image = masked ? 1 : x
// 4 consecutive pixels along W never straddle a 16-wide patch boundary.
__global__ void base_images_kernel(const float4* __restrict__ x4,
                                   const unsigned char* __restrict__ ismask,
                                   float4* __restrict__ recons,
                                   float4* __restrict__ patches_img,
                                   float4* __restrict__ mask_img)
{
    size_t i = (size_t)blockIdx.x * blockDim.x + threadIdx.x;   // over IMG_ELEMS/4
    if (i >= (size_t)(IMG_ELEMS / 4)) return;
    size_t pix = i * 4;
    int ww = pix % Wc;
    int hh = (pix / Wc) % Hc;
    int b = pix / ((size_t)Cc * Hc * Wc);
    int p = (hh / PHc) * GWc + (ww / PHc);
    float4 v = x4[i];
    recons[i] = v;
    patches_img[i] = v;
    if (ismask[b * NPc + p]) {
        float4 one = make_float4(1.0f, 1.0f, 1.0f, 1.0f);
        mask_img[i] = one;
    } else {
        mask_img[i] = v;
    }
}

// scatter pred -> recons, noise -> patches_img at masked patch pixels
__global__ void scatter_kernel(const float* __restrict__ pred,
                               const float* __restrict__ noise,
                               const int* __restrict__ shuffle,
                               float* __restrict__ recons,
                               float* __restrict__ patches_img)
{
    size_t i = (size_t)blockIdx.x * blockDim.x + threadIdx.x;
    if (i >= (size_t)Bc * NMASK * PPDc) return;
    int d = i % PPDc;
    size_t t = i / PPDc;
    int j = t % NMASK;
    int b = t / NMASK;
    int tok = shuffle[b * NPc + j];
    size_t pi = pix_index(b, tok, d);
    recons[pi] = pred[i];
    patches_img[pi] = noise[i];
}

// L1 loss: mean |pred - mask_patches| over everything
__global__ void mse_kernel(const float* __restrict__ pred,
                           const float* __restrict__ x,
                           const int* __restrict__ shuffle,
                           float* __restrict__ out)
{
    __shared__ float red[256];
    size_t i = (size_t)blockIdx.x * blockDim.x + threadIdx.x;
    float v = 0.0f;
    const size_t total = (size_t)Bc * NMASK * PPDc;
    if (i < total) {
        int d = i % PPDc;
        size_t t = i / PPDc;
        int j = t % NMASK;
        int b = t / NMASK;
        int tok = shuffle[b * NPc + j];
        v = fabsf(pred[i] - x[pix_index(b, tok, d)]);
    }
    red[threadIdx.x] = v;
    __syncthreads();
    for (int s = 128; s > 0; s >>= 1) {
        if (threadIdx.x < s) red[threadIdx.x] += red[threadIdx.x + s];
        __syncthreads();
    }
    if (threadIdx.x == 0)
        atomicAdd(out, red[0] * (1.0f / ((float)Bc * NMASK * PPDc)));
}

// ---------------------------------------------------------------------------
// Host launcher
// ---------------------------------------------------------------------------
extern "C" void kernel_launch(void* const* d_in, const int* in_sizes, int n_in,
                              void* d_out, int out_size, void* d_ws, size_t ws_size,
                              hipStream_t stream) {
    const float* x        = (const float*)d_in[0];
    const int*   shuffle  = (const int*)  d_in[1];
    const float* W_pe     = (const float*)d_in[2];
    const float* b_pe     = (const float*)d_in[3];
    const float* pos      = (const float*)d_in[4];
    const float* enc_W    = (const float*)d_in[5];
    const float* enc_b    = (const float*)d_in[6];
    const float* in_proj_w= (const float*)d_in[7];
    const float* in_proj_b= (const float*)d_in[8];
    const float* out_w    = (const float*)d_in[9];
    const float* out_b    = (const float*)d_in[10];
    const float* head_W   = (const float*)d_in[11];
    const float* head_b   = (const float*)d_in[12];
    const float* noise    = (const float*)d_in[13];

    float* out_recons  = (float*)d_out;
    float* out_patches = out_recons + IMG_ELEMS;
    float* out_mse     = out_patches + IMG_ELEMS;
    float* out_maskimg = out_mse + 1;

    char* ws = (char*)d_ws;
    size_t off = 0;
    auto alloc = [&](size_t bytes) -> char* {
        char* p = ws + off;
        off = (off + bytes + 255) & ~(size_t)255;
        return p;
    };

    _Float16* wpe16   = (_Float16*)alloc((size_t)Ec * PPDc * 2);
    _Float16* encW16  = (_Float16*)alloc((size_t)Ec * Ec * 2);
    _Float16* inpj16  = (_Float16*)alloc((size_t)3 * Ec * Ec * 2);
    _Float16* outw16  = (_Float16*)alloc((size_t)Ec * Ec * 2);
    _Float16* headW16 = (_Float16*)alloc((size_t)PPDc * Ec * 2);
    _Float16* maskp16 = (_Float16*)alloc((size_t)Bc * NMASK * PPDc * 2);
    _Float16* unmkp16 = (_Float16*)alloc((size_t)Bc * NUNM * PPDc * 2);
    _Float16* umtok16 = (_Float16*)alloc((size_t)Bc * NUNM * Ec * 2);
    _Float16* mtok16  = (_Float16*)alloc((size_t)Bc * NMASK * Ec * 2);
    _Float16* enc16   = (_Float16*)alloc((size_t)Bc * NUNM * Ec * 2);
    _Float16* q16     = (_Float16*)alloc((size_t)Bc * NMASK * Ec * 2);
    _Float16* k16     = (_Float16*)alloc((size_t)Bc * NUNM * Ec * 2);
    _Float16* vT16    = (_Float16*)alloc((size_t)Bc * NHc * HDc * 64 * 2);
    _Float16* ao16    = (_Float16*)alloc((size_t)Bc * NMASK * Ec * 2);
    _Float16* dec16   = (_Float16*)alloc((size_t)Bc * NMASK * Ec * 2);
    float*    pred32  = (float*)   alloc((size_t)Bc * NMASK * PPDc * 4);
    unsigned char* ismask = (unsigned char*)alloc((size_t)Bc * NPc);

    auto blks = [](size_t n, int t) { return (int)((n + t - 1) / t); };

    // weight conversions
    f32_to_f16_kernel<<<blks(Ec*PPDc,256),256,0,stream>>>(W_pe, wpe16, Ec*PPDc);
    f32_to_f16_kernel<<<blks(Ec*Ec,256),256,0,stream>>>(enc_W, encW16, Ec*Ec);
    f32_to_f16_kernel<<<blks(3*Ec*Ec,256),256,0,stream>>>(in_proj_w, inpj16, 3*Ec*Ec);
    f32_to_f16_kernel<<<blks(Ec*Ec,256),256,0,stream>>>(out_w, outw16, Ec*Ec);
    f32_to_f16_kernel<<<blks(PPDc*Ec,256),256,0,stream>>>(head_W, headW16, PPDc*Ec);

    // mask bitmap + zero-padded vT + mse accumulator
    zero_mask_kernel<<<blks(Bc*NPc,256),256,0,stream>>>(ismask, Bc*NPc);
    set_mask_kernel<<<blks(Bc*NMASK,256),256,0,stream>>>(shuffle, ismask);
    zero_f16_kernel<<<blks((size_t)Bc*NHc*HDc*64,256),256,0,stream>>>(vT16, Bc*NHc*HDc*64);
    zero_f32_kernel<<<1,32,0,stream>>>(out_mse, 1);

    // gather patches (f16) + base image outputs
    gather_patches_kernel<<<blks((size_t)Bc*NPc*PPDc,256),256,0,stream>>>(x, shuffle, maskp16, unmkp16);
    base_images_kernel<<<blks((size_t)IMG_ELEMS/4,256),256,0,stream>>>(
        (const float4*)x, ismask, (float4*)out_recons, (float4*)out_patches, (float4*)out_maskimg);

    const int Mu = Bc * NUNM;   // 3136
    const int Mm = Bc * NMASK;  // 9408
    auto gblocks = [](int M, int N) { return ((M / 32) * (N / 64) + 7) / 8; };

    // T1: unmask_tokens = unmask_patches @ W_pe^T + b_pe + pe[unmask_ind]
    gemm_wmma<1><<<gblocks(Mu,Ec),256,0,stream>>>(unmkp16, wpe16, b_pe, Mu, Ec, PPDc,
        umtok16, nullptr, pos, shuffle, NUNM, NMASK, nullptr, nullptr);
    // T2: encoded = gelu(unmask_tokens @ enc_W^T + enc_b)
    gemm_wmma<2><<<gblocks(Mu,Ec),256,0,stream>>>(umtok16, encW16, enc_b, Mu, Ec, Ec,
        enc16, nullptr, nullptr, nullptr, 0, 0, nullptr, nullptr);
    // T3: mask_tokens = mask_patches @ W_pe^T + b_pe + pe[mask_ind]
    gemm_wmma<1><<<gblocks(Mm,Ec),256,0,stream>>>(maskp16, wpe16, b_pe, Mm, Ec, PPDc,
        mtok16, nullptr, pos, shuffle, NMASK, 0, nullptr, nullptr);
    // q = mask_tokens @ Wq^T + bq
    gemm_wmma<0><<<gblocks(Mm,Ec),256,0,stream>>>(mtok16, inpj16, in_proj_b, Mm, Ec, Ec,
        q16, nullptr, nullptr, nullptr, 0, 0, nullptr, nullptr);
    // k,v fused: encoded @ [Wk;Wv]^T + [bk;bv]; v written transposed per head
    gemm_wmma<4><<<gblocks(Mu,2*Ec),256,0,stream>>>(enc16, inpj16 + (size_t)Ec*Ec,
        in_proj_b + Ec, Mu, 2*Ec, Ec,
        nullptr, nullptr, nullptr, nullptr, 0, 0, k16, vT16);
    // attention
    attention_kernel<<<Bc*NHc,256,0,stream>>>(q16, k16, vT16, ao16);
    // out projection
    gemm_wmma<0><<<gblocks(Mm,Ec),256,0,stream>>>(ao16, outw16, out_b, Mm, Ec, Ec,
        dec16, nullptr, nullptr, nullptr, 0, 0, nullptr, nullptr);
    // decoder head -> pred (f32)
    gemm_wmma<3><<<gblocks(Mm,PPDc),256,0,stream>>>(dec16, headW16, head_b, Mm, PPDc, Ec,
        nullptr, pred32, nullptr, nullptr, 0, 0, nullptr, nullptr);

    // loss + scatter pred/noise into image outputs
    mse_kernel<<<blks((size_t)Mm*PPDc,256),256,0,stream>>>(pred32, x, shuffle, out_mse);
    scatter_kernel<<<blks((size_t)Mm*PPDc,256),256,0,stream>>>(pred32, noise, shuffle, out_recons, out_patches);
}
